// AttentionBlock_69217692942510
// MI455X (gfx1250) — matrix operand order repaired and checked by reading, not compiled, not run on hardware
//
#include <hip/hip_runtime.h>

// ---------- types ----------
typedef unsigned short u16;
typedef unsigned int   u32;
typedef __attribute__((ext_vector_type(16))) __bf16 bf16x16;
typedef __attribute__((ext_vector_type(8)))  float  f32x8;
typedef __attribute__((ext_vector_type(4)))  u32    u32x4;
typedef __attribute__((ext_vector_type(4)))  float  f32x4;
typedef __attribute__((ext_vector_type(4)))  int    i32x4;
typedef __attribute__((address_space(1))) i32x4 gbl_i32x4;   // global (AS1)
typedef __attribute__((address_space(3))) i32x4 lds_i32x4;   // LDS (AS3)

#define B_   2
#define C_   512
#define S_   4096
#define H_   8
#define D_   64
#define OC3  1536
#define SCALE_ 0.125f   // HEAD_DIM^-0.5 = 64^-0.5

union Frag { bf16x16 v; u32x4 q[2]; };

__device__ __forceinline__ u16 f2bf(float x) {
  union { float f; u32 u; } t; t.f = x;
  u32 u = t.u;
  if ((u & 0x7f800000u) != 0x7f800000u)           // round-to-nearest-even, keep inf/nan
    u = u + 0x7fffu + ((u >> 16) & 1u);
  return (u16)(u >> 16);
}

// ---- async 16B copy global -> LDS (gfx1250, ASYNCcnt-tracked) ----
// generic LDS pointer low 32 bits == DS offset (aperture rule); integer
// round-trip builds the AS(3) pointer and discards const for the AS(1) side.
__device__ __forceinline__ void async_cp16(const u16* g, u16* l) {
  __builtin_amdgcn_global_load_async_to_lds_b128(
      (gbl_i32x4*)(unsigned long long)(size_t)g,
      (lds_i32x4*)(unsigned int)(size_t)l,
      0, 0);
}

// A-matrix fragment (16x32 bf16): lane r=lane%16 holds row r.
// element e<8  -> K = kbase + half*8 + e          (contiguous 16B)
// element e>=8 -> K = kbase + 16 + half*8 + (e-8) (contiguous 16B)
__device__ __forceinline__ bf16x16 load_a(const u16* row, int kbase, int half) {
  Frag f;
  f.q[0] = *(const u32x4*)(row + kbase + half * 8);
  f.q[1] = *(const u32x4*)(row + kbase + 16 + half * 8);
  return f.v;
}

// B-matrix fragment (32x16 bf16): lane n holds column n; element e<8 -> K = half*8 + e, e>=8 -> K = 16 + half*8 + (e-8)
__device__ __forceinline__ bf16x16 load_b(const u16* colKrun, int half) {
  Frag f;
  f.q[0] = *(const u32x4*)(colKrun + half * 8);
  f.q[1] = *(const u32x4*)(colKrun + 16 + half * 8);
  return f.v;
}

__device__ __forceinline__ f32x8 wmma_bf16(bf16x16 a, bf16x16 b, f32x8 c) {
  return __builtin_amdgcn_wmma_f32_16x16x32_bf16(false, a, false, b, (short)0, c, false, false);
}

// ---------- kernel 0a: fp32 -> bf16 weight convert ----------
__global__ void cvt_bf16(const float* __restrict__ src, u16* __restrict__ dst, int n) {
  int i = blockIdx.x * blockDim.x + threadIdx.x;
  int stride = gridDim.x * blockDim.x;
  for (; i < n; i += stride) dst[i] = f2bf(src[i]);
}

// ---------- kernel 0b: x [b,c,s] fp32 -> Xb [b,s,c] bf16 (LDS tile transpose) ----------
__global__ void xpose(const float* __restrict__ x, u16* __restrict__ Xb) {
  __shared__ float t[32][33];
  const int s0 = blockIdx.x * 32, c0 = blockIdx.y * 32, b = blockIdx.z;
  const int tx = threadIdx.x, ty = threadIdx.y;
  const float* src = x + ((size_t)b * C_ + c0) * S_ + s0;
#pragma unroll
  for (int j = 0; j < 32; j += 8)
    t[ty + j][tx] = src[(size_t)(ty + j) * S_ + tx];
  __syncthreads();
  u16* dst = Xb + ((size_t)b * S_ + s0) * C_ + c0;
#pragma unroll
  for (int j = 0; j < 32; j += 8)
    dst[(size_t)(ty + j) * C_ + tx] = f2bf(t[tx][ty + j]);
}

// ---------- kernel 1: QKV GEMM with async double-buffered weight staging ----------
// Qh,Kh: [b,H,s,d] bf16 ; Vt: [b,H,d,s] bf16 (transposed for PV B-fragments)
__global__ void __launch_bounds__(256) qkv_gemm(const u16* __restrict__ Xb,
                                                const u16* __restrict__ Wq,
                                                u16* __restrict__ Qh,
                                                u16* __restrict__ Kh,
                                                u16* __restrict__ Vt) {
  __shared__ u16 Wt[2][128 * 32];     // 128 out-cols x 32 k, double buffered (16 KB)
  const int s0 = blockIdx.x * 128;
  const int o0 = blockIdx.y * 128;
  const int b  = blockIdx.z;
  const int lane = threadIdx.x & 31;
  const int wave = threadIdx.x >> 5;
  const int half = lane >> 4;
  const int nLo  = lane & 15;
  const u16* arow = Xb + ((size_t)b * S_ + (s0 + wave * 16 + nLo)) * C_;

  auto issueW = [&](int k, int buf) {
#pragma unroll
    for (int c = 0; c < 2; ++c) {
      int chunk = threadIdx.x + c * 256;   // 512 x 16B chunks = 8 KB tile
      int row = chunk >> 2;                // 4 chunks per 32-k row
      int off = (chunk & 3) * 8;
      async_cp16(Wq + (size_t)(o0 + row) * C_ + k + off, &Wt[buf][row * 32 + off]);
    }
  };

  f32x8 acc[8] = {};
  issueW(0, 0);
  for (int k = 0, step = 0; k < C_; k += 32, ++step) {
    const int cur = step & 1;
    if (k + 32 < C_) {
      issueW(k + 32, cur ^ 1);
      __builtin_amdgcn_s_wait_asynccnt(2);   // tile `cur` done (in-order completion)
    } else {
      __builtin_amdgcn_s_wait_asynccnt(0);
    }
    __syncthreads();
    bf16x16 a = load_a(arow, k, half);
#pragma unroll
    for (int nt = 0; nt < 8; ++nt)
      acc[nt] = wmma_bf16(a, load_b(&Wt[cur][(nt * 16 + nLo) * 32], half), acc[nt]);
    __syncthreads();                         // buffer reuse guard
  }
#pragma unroll
  for (int nt = 0; nt < 8; ++nt) {
    int o     = o0 + nt * 16 + nLo;
    int which = o >> 9;          // 0=Q 1=K 2=V (128-col tiles never straddle)
    int oo    = o & 511;
    int h = oo >> 6, dd = oo & 63;
    int rowBase = s0 + wave * 16 + half * 8; // C-layout: VGPR i -> row half*8+i
    if (which == 2) {
      u16* dst = Vt + (((size_t)b * H_ + h) * D_ + dd) * S_ + rowBase;
#pragma unroll
      for (int i = 0; i < 8; ++i) dst[i] = f2bf(acc[nt][i]);
    } else {
      u16* base = (which == 0) ? Qh : Kh;
      size_t rb = (((size_t)b * H_ + h) * S_ + rowBase) * D_ + dd;
#pragma unroll
      for (int i = 0; i < 8; ++i) base[rb + (size_t)i * D_] = f2bf(acc[nt][i]);
    }
  }
}

// ---------- kernel 2: flash attention, async double-buffered K/V staging ----------
__global__ void __launch_bounds__(256) attn(const u16* __restrict__ Qh,
                                            const u16* __restrict__ Kh,
                                            const u16* __restrict__ Vt,
                                            u16* __restrict__ AOut) {
  __shared__ u16 Kt[2][64 * 64];     // k-tile: 64 keys x 64 d   (16 KB dbuf)
  __shared__ u16 Vts[2][64 * 64];    // v-tile: 64 d x 64 keys   (16 KB dbuf)
  __shared__ u16 ldsP[8][16 * 64];   // per-wave P tile (16 KB)
  const int q0 = blockIdx.x * 128;
  const int h  = blockIdx.y;
  const int b  = blockIdx.z;
  const int lane = threadIdx.x & 31;
  const int wave = threadIdx.x >> 5;
  const int half = lane >> 4;
  const int nLo  = lane & 15;
  const size_t bh = (size_t)b * H_ + h;

  auto issueKV = [&](int kt, int buf) {
#pragma unroll
    for (int c = 0; c < 2; ++c) {
      int chunk = threadIdx.x + c * 256;   // 512 x 16B chunks per 8 KB tile
      int row = chunk >> 3;                // 8 chunks per 64-elem row
      int off = (chunk & 7) * 8;
      async_cp16(Kh + (bh * S_ + (size_t)(kt * 64 + row)) * D_ + off,
                 &Kt[buf][row * 64 + off]);
      async_cp16(Vt + (bh * D_ + row) * S_ + kt * 64 + off,
                 &Vts[buf][row * 64 + off]);
    }
  };

  const u16* qptr = Qh + (bh * S_ + (q0 + wave * 16 + nLo)) * D_;
  const bf16x16 aQ0 = load_a(qptr, 0, half);
  const bf16x16 aQ1 = load_a(qptr, 32, half);

  f32x8 acc[4] = {};
  float m[8], l[8];
#pragma unroll
  for (int i = 0; i < 8; ++i) { m[i] = -1e30f; l[i] = 0.f; }

  u16* myP = &ldsP[wave][0];

  issueKV(0, 0);
  for (int kt = 0; kt < S_ / 64; ++kt) {
    const int cur = kt & 1;
    if (kt + 1 < S_ / 64) {
      issueKV(kt + 1, cur ^ 1);
      __builtin_amdgcn_s_wait_asynccnt(4);  // tile `cur` (4 issues/lane) complete
    } else {
      __builtin_amdgcn_s_wait_asynccnt(0);
    }
    __syncthreads();

    // ---- S = (Q K^T) * scale, 16x64 tile, B-fragments from LDS ----
    f32x8 sacc[4] = {};
#pragma unroll
    for (int nt = 0; nt < 4; ++nt) {
      const u16* krow = &Kt[cur][(nt * 16 + nLo) * 64];
      sacc[nt] = wmma_bf16(aQ0, load_b(krow, half), sacc[nt]);
      sacc[nt] = wmma_bf16(aQ1, load_b(krow + 32, half), sacc[nt]);
    }
    // ---- online softmax: row max across 16 lanes of each half ----
#pragma unroll
    for (int i = 0; i < 8; ++i) {
      float r = -1e30f;
#pragma unroll
      for (int nt = 0; nt < 4; ++nt) { sacc[nt][i] *= SCALE_; r = fmaxf(r, sacc[nt][i]); }
      r = fmaxf(r, __shfl_xor(r, 1));
      r = fmaxf(r, __shfl_xor(r, 2));
      r = fmaxf(r, __shfl_xor(r, 4));
      r = fmaxf(r, __shfl_xor(r, 8));
      float mnew = fmaxf(m[i], r);
      float alpha = __expf(m[i] - mnew);
      m[i] = mnew;
      l[i] *= alpha;
#pragma unroll
      for (int dn = 0; dn < 4; ++dn) acc[dn][i] *= alpha;
    }
    // ---- P = exp(S-m): row-sum + spill to LDS as bf16 (C-layout -> A-layout) ----
#pragma unroll
    for (int i = 0; i < 8; ++i) {
      float rs = 0.f;
#pragma unroll
      for (int nt = 0; nt < 4; ++nt) {
        float p = __expf(sacc[nt][i] - m[i]);
        rs += p;
        myP[(half * 8 + i) * 64 + nt * 16 + nLo] = f2bf(p);
      }
      rs += __shfl_xor(rs, 1);
      rs += __shfl_xor(rs, 2);
      rs += __shfl_xor(rs, 4);
      rs += __shfl_xor(rs, 8);
      l[i] += rs;
    }
    // ---- O += P * V : A-fragments from P-LDS, B-fragments from V-LDS ----
    const u16* prow = myP + nLo * 64;
    const bf16x16 aP0 = load_a(prow, 0, half);
    const bf16x16 aP1 = load_a(prow, 32, half);
#pragma unroll
    for (int dn = 0; dn < 4; ++dn) {
      const u16* vrow = &Vts[cur][(dn * 16 + nLo) * 64];
      acc[dn] = wmma_bf16(aP0, load_b(vrow, half), acc[dn]);
      acc[dn] = wmma_bf16(aP1, load_b(vrow + 32, half), acc[dn]);
    }
    __syncthreads();                        // buffer reuse guard
  }
  // ---- normalize, store AOut[b,s,c] bf16 ----
#pragma unroll
  for (int i = 0; i < 8; ++i) {
    float inv = 1.0f / l[i];
    int row = q0 + wave * 16 + half * 8 + i;
    u16* dst = AOut + ((size_t)b * S_ + row) * C_ + h * D_;
#pragma unroll
    for (int dn = 0; dn < 4; ++dn) dst[dn * 16 + nLo] = f2bf(acc[dn][i] * inv);
  }
}

// ---------- kernel 3: out-projection + bias, async weight staging ----------
__global__ void __launch_bounds__(256) out_proj(const u16* __restrict__ AOut,
                                                const u16* __restrict__ Wo,
                                                const float* __restrict__ bias,
                                                float* __restrict__ Out) {
  __shared__ u16 Wt[2][128 * 32];
  const int s0 = blockIdx.x * 128;
  const int o0 = blockIdx.y * 128;
  const int b  = blockIdx.z;
  const int lane = threadIdx.x & 31;
  const int wave = threadIdx.x >> 5;
  const int half = lane >> 4;
  const int nLo  = lane & 15;
  const u16* arow = AOut + ((size_t)b * S_ + (s0 + wave * 16 + nLo)) * C_;

  auto issueW = [&](int k, int buf) {
#pragma unroll
    for (int c = 0; c < 2; ++c) {
      int chunk = threadIdx.x + c * 256;
      int row = chunk >> 2;
      int off = (chunk & 3) * 8;
      async_cp16(Wo + (size_t)(o0 + row) * C_ + k + off, &Wt[buf][row * 32 + off]);
    }
  };

  f32x8 acc[8] = {};
  issueW(0, 0);
  for (int k = 0, step = 0; k < C_; k += 32, ++step) {
    const int cur = step & 1;
    if (k + 32 < C_) {
      issueW(k + 32, cur ^ 1);
      __builtin_amdgcn_s_wait_asynccnt(2);
    } else {
      __builtin_amdgcn_s_wait_asynccnt(0);
    }
    __syncthreads();
    bf16x16 a = load_a(arow, k, half);
#pragma unroll
    for (int nt = 0; nt < 8; ++nt)
      acc[nt] = wmma_bf16(a, load_b(&Wt[cur][(nt * 16 + nLo) * 32], half), acc[nt]);
    __syncthreads();
  }
#pragma unroll
  for (int nt = 0; nt < 8; ++nt) {
    int o = o0 + nt * 16 + nLo;
    float bb = bias[o];
    float* dst = Out + ((size_t)b * C_ + o) * S_ + s0 + wave * 16 + half * 8;
    f32x4 v0 = { acc[nt][0] + bb, acc[nt][1] + bb, acc[nt][2] + bb, acc[nt][3] + bb };
    f32x4 v1 = { acc[nt][4] + bb, acc[nt][5] + bb, acc[nt][6] + bb, acc[nt][7] + bb };
    *(f32x4*)(dst)     = v0;
    *(f32x4*)(dst + 4) = v1;
  }
}

// ---------- host ----------
extern "C" void kernel_launch(void* const* d_in, const int* in_sizes, int n_in,
                              void* d_out, int out_size, void* d_ws, size_t ws_size,
                              hipStream_t stream) {
  (void)in_sizes; (void)n_in; (void)out_size; (void)ws_size;
  const float* x     = (const float*)d_in[0];
  const float* w_qkv = (const float*)d_in[1];
  const float* w_out = (const float*)d_in[2];
  const float* b_out = (const float*)d_in[3];
  float* out = (float*)d_out;

  char* ws = (char*)d_ws;
  size_t off = 0;
  auto alloc = [&](size_t bytes) -> void* {
    void* p = ws + off;
    off += bytes;
    off = (off + 255) & ~(size_t)255;
    return p;
  };
  u16* Xb = (u16*)alloc((size_t)B_ * S_ * C_ * 2);        // x transposed, bf16
  u16* Wq = (u16*)alloc((size_t)OC3 * C_ * 2);            // w_qkv bf16
  u16* Wo = (u16*)alloc((size_t)C_ * C_ * 2);             // w_out bf16
  u16* Qh = (u16*)alloc((size_t)B_ * H_ * S_ * D_ * 2);   // [b,H,s,d]
  u16* Kh = (u16*)alloc((size_t)B_ * H_ * S_ * D_ * 2);   // [b,H,s,d]
  u16* Vt = (u16*)alloc((size_t)B_ * H_ * D_ * S_ * 2);   // [b,H,d,s]
  u16* AO = (u16*)alloc((size_t)B_ * S_ * C_ * 2);        // attention out [b,s,c]

  cvt_bf16<<<512, 256, 0, stream>>>(w_qkv, Wq, OC3 * C_);
  cvt_bf16<<<256, 256, 0, stream>>>(w_out, Wo, C_ * C_);
  xpose<<<dim3(S_ / 32, C_ / 32, B_), dim3(32, 8), 0, stream>>>(x, Xb);
  qkv_gemm<<<dim3(S_ / 128, OC3 / 128, B_), 256, 0, stream>>>(Xb, Wq, Qh, Kh, Vt);
  attn<<<dim3(S_ / 128, H_, B_), 256, 0, stream>>>(Qh, Kh, Vt, AO);
  out_proj<<<dim3(S_ / 128, C_ / 128, B_), 256, 0, stream>>>(AO, Wo, b_out, out);
}
